// LIFNeurons_44032004719190
// MI455X (gfx1250) — matrix operand order repaired and checked by reading, not compiled, not run on hardware
//
#include <hip/hip_runtime.h>

typedef __attribute__((ext_vector_type(16))) __bf16 v16bf;
typedef __attribute__((ext_vector_type(8)))  __bf16 v8bf;
typedef __attribute__((ext_vector_type(8)))  float  v8f;
typedef int v4i_vec __attribute__((vector_size(16)));   // matches async builtin param

#define N_IN    2048
#define N_OUT   2048
#define T_TOT   512
#define BATCH   32

#define T_CHUNK 32      // timesteps of currents computed per GEMM phase
#define O_TILE  128     // output neurons per block
#define K_STEP  64      // K per barrier (2 x 32-K WMMA sub-steps)
#define THREADS 256     // 8 waves (wave32)

#if __has_builtin(__builtin_amdgcn_global_load_async_to_lds_b128)
#define HAVE_ASYNC_LDS 1
#endif

// Copy 16 bytes global(bf16) -> LDS. Async (ASYNCcnt) when available.
__device__ __forceinline__ void copy16_g2l(const __bf16* gsrc, __bf16* ldst) {
#ifdef HAVE_ASYNC_LDS
    __builtin_amdgcn_global_load_async_to_lds_b128(
        (__attribute__((address_space(1))) v4i_vec*)gsrc,
        (__attribute__((address_space(3))) v4i_vec*)ldst, 0, 0);
#else
    *(float4*)ldst = *(const float4*)gsrc;
#endif
}

__device__ __forceinline__ void wait_async_lds() {
#ifdef HAVE_ASYNC_LDS
#if __has_builtin(__builtin_amdgcn_s_wait_asynccnt)
    __builtin_amdgcn_s_wait_asynccnt(0);
#else
    asm volatile("s_wait_asynccnt 0x0" ::: "memory");
#endif
#endif
}

// ---- one-shot: W f32 [N_OUT, N_IN] -> bf16 in workspace ----
__global__ __launch_bounds__(256)
void convert_w_bf16(const float* __restrict__ W, __bf16* __restrict__ Wb) {
    const size_t i = ((size_t)blockIdx.x * 256 + threadIdx.x) * 8;
    const float4 a = *(const float4*)(W + i);
    const float4 b = *(const float4*)(W + i + 4);
    v8bf o;
    o[0] = (__bf16)a.x; o[1] = (__bf16)a.y; o[2] = (__bf16)a.z; o[3] = (__bf16)a.w;
    o[4] = (__bf16)b.x; o[5] = (__bf16)b.y; o[6] = (__bf16)b.z; o[7] = (__bf16)b.w;
    *(v8bf*)(Wb + i) = o;
}

// ---- one-shot: pre f32 [B][N_IN][T] -> bf16 [B][T][N_IN] (transpose) ----
__global__ __launch_bounds__(256)
void convert_pre_bf16t(const float* __restrict__ pre, __bf16* __restrict__ preb) {
    __shared__ __bf16 tile[32 * 33];   // +1 pad: conflict-free transposed access

    const int b  = blockIdx.z;
    const int k0 = blockIdx.y * 32;
    const int t0 = blockIdx.x * 32;
    const int tid = threadIdx.x;

    // read: 32 k-rows x 32 t-cols, float4 per thread (coalesced along t)
    {
        const int row = tid >> 3;           // k row 0..31
        const int tc  = (tid & 7) * 4;      // t col
        const float4 v = *(const float4*)(pre +
            ((size_t)b * N_IN + k0 + row) * T_TOT + t0 + tc);
        tile[(tc + 0) * 33 + row] = (__bf16)v.x;
        tile[(tc + 1) * 33 + row] = (__bf16)v.y;
        tile[(tc + 2) * 33 + row] = (__bf16)v.z;
        tile[(tc + 3) * 33 + row] = (__bf16)v.w;
    }
    __syncthreads();

    // write: 32 t-rows x 32 k-cols bf16 (coalesced along k)
    {
        const int trow = tid >> 3;          // t row 0..31
        const int kc   = (tid & 7) * 4;     // k col
        __bf16* dst = preb + ((size_t)b * T_TOT + t0 + trow) * N_IN + k0 + kc;
        const __bf16* src = &tile[trow * 33 + kc];
        dst[0] = src[0]; dst[1] = src[1]; dst[2] = src[2]; dst[3] = src[3];
    }
}

__global__ __launch_bounds__(THREADS)
void lif_fused_wmma(const __bf16* __restrict__ preb,     // [B, T, N_IN] bf16
                    const __bf16* __restrict__ Wb,       // [N_OUT, N_IN] bf16
                    const float*  __restrict__ bias,     // [N_OUT]
                    const float*  __restrict__ beta_syn, // [N_OUT]
                    const float*  __restrict__ beta_mem, // [N_OUT]
                    float* __restrict__ out)             // [B, N_OUT, T]
{
    __shared__ __bf16 A_lds[2][T_CHUNK * K_STEP];    // [t][k], 2 * 4 KB
    __shared__ __bf16 B_lds[2][O_TILE * K_STEP];     // [o][k], 2 * 16 KB
    __shared__ float  curr[T_CHUNK * O_TILE];        // 16 KB
    __shared__ float  spk [O_TILE * T_CHUNK];        // 16 KB

    const int tid    = threadIdx.x;
    const int lane   = tid & 31;
    const int wave   = tid >> 5;
    const int wave_t = wave & 1;    // which 16-row t slab
    const int wave_o = wave >> 1;   // which 32-col o slab
    const int half   = lane >> 4;
    const int l15    = lane & 15;

    const int bb     = blockIdx.y;
    const int o_base = blockIdx.x * O_TILE;

    const size_t pre_bt = (size_t)bb * T_TOT * N_IN;   // base of preb[b]

    // ---- persistent LIF state (one output neuron per scan thread) ----
    float syn = 0.f, mem = 0.f, bsc = 0.f, bmc = 0.f, bia = 0.f;
    if (tid < O_TILE) {
        const int o = o_base + tid;
        bia = bias[o];
        bsc = fminf(fmaxf(beta_syn[o], 0.f), 1.f);
        bmc = fminf(fmaxf(beta_mem[o], 0.f), 1.f);
    }

    // A-tile loader: 32 t-rows x 64 k (bf16) = 4 KB = 256 x 16B, 1/thread.
    const int at_row = tid >> 3;         // t row 0..31
    const int at_c   = (tid & 7) * 8;    // bf16 col 0..56

    // B-tile loader: 128 o-rows x 64 k (bf16) = 16 KB = 1024 x 16B, 4/thread.
    const int b_row  = tid >> 1;         // o row 0..127
    const int b_col0 = (tid & 1) * 32;   // starting bf16 col: 0 or 32

    const int KITERS = N_IN / K_STEP;   // 32

    for (int chunk = 0; chunk < T_TOT / T_CHUNK; ++chunk) {
        const int t0 = chunk * T_CHUNK;

        v8f c0 = {};
        v8f c1 = {};

        // ---- preload tiles for k-iter 0 (all async b128, uniform) ----
        {
            copy16_g2l(preb + pre_bt + (size_t)(t0 + at_row) * N_IN + at_c,
                       A_lds[0] + at_row * K_STEP + at_c);

            const __bf16* g = Wb + (size_t)(o_base + b_row) * N_IN + b_col0;
            __bf16*       l = B_lds[0] + b_row * K_STEP + b_col0;
            copy16_g2l(g,      l);
            copy16_g2l(g +  8, l +  8);
            copy16_g2l(g + 16, l + 16);
            copy16_g2l(g + 24, l + 24);
        }
        wait_async_lds();
        __syncthreads();

        for (int it = 0; it < KITERS; ++it) {
            const int k0  = it * K_STEP;
            const int cur = it & 1;

            // ---- prefetch next tiles into the other buffers ----
            if (it + 1 < KITERS) {
                const int kn = k0 + K_STEP;
                copy16_g2l(preb + pre_bt + (size_t)(t0 + at_row) * N_IN + kn + at_c,
                           A_lds[cur ^ 1] + at_row * K_STEP + at_c);

                const __bf16* g = Wb + (size_t)(o_base + b_row) * N_IN + kn + b_col0;
                __bf16*       l = B_lds[cur ^ 1] + b_row * K_STEP + b_col0;
                copy16_g2l(g,      l);
                copy16_g2l(g +  8, l +  8);
                copy16_g2l(g + 16, l + 16);
                copy16_g2l(g + 24, l + 24);
            }

            // ---- two 32-K WMMA sub-steps per barrier ----
            const __bf16* asrc = A_lds[cur] + (wave_t * 16 + l15) * K_STEP;
            const __bf16* bs   = B_lds[cur] + (wave_o * 32 + l15) * K_STEP + 16 * half;

            #pragma unroll
            for (int s = 0; s < 2; ++s) {
                // A fragment: 16x32 bf16, 16 contiguous k per lane
                v16bf af;
                #pragma unroll
                for (int v = 0; v < 8; ++v) {
                    const int kb = s * 32 + 2 * v + 16 * half;
                    af[2 * v]     = asrc[kb];
                    af[2 * v + 1] = asrc[kb + 1];
                }
                // B fragments: 16 contiguous k per lane
                v16bf bf0, bf1;
                #pragma unroll
                for (int j = 0; j < 16; ++j) {
                    bf0[j] = bs[s * 32 + j];
                    bf1[j] = bs[16 * K_STEP + s * 32 + j];
                }
                c0 = __builtin_amdgcn_wmma_f32_16x16x32_bf16(false, af, false, bf0,
                                                             (short)0, c0, false, false);
                c1 = __builtin_amdgcn_wmma_f32_16x16x32_bf16(false, af, false, bf1,
                                                             (short)0, c1, false, false);
            }

            wait_async_lds();
            __syncthreads();   // double-buffer handoff
        }

        // ---- spill C fragments (currents) to LDS as [t][o] ----
        #pragma unroll
        for (int v = 0; v < 8; ++v) {
            const int m  = wave_t * 16 + half * 8 + v;
            const int n0 = wave_o * 32 + l15;
            curr[m * O_TILE + n0]      = c0[v];
            curr[m * O_TILE + n0 + 16] = c1[v];
        }
        __syncthreads();

        // ---- sequential LIF scan over this chunk (state in registers) ----
        if (tid < O_TILE) {
            #pragma unroll 4
            for (int t = 0; t < T_CHUNK; ++t) {
                const float I = curr[t * O_TILE + tid] + bia;
                syn = syn * bsc + I;
                mem = mem * bmc + syn * (1.f - bmc);
                const float s = (mem - 1.f) > 0.f ? 1.f : 0.f;
                mem -= s;   // soft reset (detached in reference)
                spk[tid * T_CHUNK + t] = s;
            }
        }
        __syncthreads();

        // ---- coalesced writeback: 32-lane groups write contiguous 128 B runs ----
        #pragma unroll
        for (int i = 0; i < (O_TILE * T_CHUNK) / THREADS; ++i) {
            const int idx = i * THREADS + tid;
            const int ol  = idx >> 5;
            const int tl  = idx & 31;
            out[((size_t)bb * N_OUT + o_base + ol) * T_TOT + t0 + tl] =
                spk[ol * T_CHUNK + tl];
        }
    }
}

extern "C" void kernel_launch(void* const* d_in, const int* in_sizes, int n_in,
                              void* d_out, int out_size, void* d_ws, size_t ws_size,
                              hipStream_t stream) {
    const float* pre      = (const float*)d_in[0];  // [B, N_IN, T]
    const float* W        = (const float*)d_in[1];  // [N_OUT, N_IN]
    const float* bias     = (const float*)d_in[2];  // [N_OUT]
    const float* beta_syn = (const float*)d_in[3];  // [N_OUT]
    const float* beta_mem = (const float*)d_in[4];  // [N_OUT]
    float* out            = (float*)d_out;          // [B, N_OUT, T]

    __bf16* Wb   = (__bf16*)d_ws;                           // 8 MiB
    __bf16* preb = (__bf16*)((char*)d_ws + (size_t)N_OUT * N_IN * 2); // 64 MiB

    convert_w_bf16<<<(N_OUT * (size_t)N_IN) / (256 * 8), 256, 0, stream>>>(W, Wb);

    dim3 cgrid(T_TOT / 32, N_IN / 32, BATCH);   // (16, 64, 32)
    convert_pre_bf16t<<<cgrid, 256, 0, stream>>>(pre, preb);

    dim3 grid(N_OUT / O_TILE, BATCH);   // (16, 32)
    dim3 block(THREADS);
    lif_fused_wmma<<<grid, block, 0, stream>>>(preb, Wb, bias, beta_syn, beta_mem, out);
}
